// GrainNN_classifier_36636071035479
// MI455X (gfx1250) — compile-verified, matrix operand order checked by reading, not run on hardware
//
#include <hip/hip_runtime.h>

// ---------------------------------------------------------------------------
// GrainNN hetero-GCLSTM for MI455X (gfx1250, wave32, WMMA).
//
// Pipeline per cell (4 cells: enc0, enc1, dec0, dec1):
//   1. segment-sum scatters (3 edge types) -> mean accumulators  (fp32 atomics)
//   2. build A matrices in WMMA-fragment bf16 layout:
//        A_joint = [mean_jj | mean_gj | x_in | h]   (K = 3*IN + 64, padded /32)
//        A_grain = [mean_pg | x_in | h]             (K = 2*IN + 64, padded /32)
//   3. pack fused gate weights W (K x 256, gate blocks i|f|c|o) into fragment
//      layout + fused bias
//   4. fused WMMA GEMM (v_wmma_f32_16x16x32_bf16) + LSTM gate math, updating
//      (h, c) state in place.
// Head: per-jj-edge dot products with lin1/lin2.
// ---------------------------------------------------------------------------

typedef __attribute__((ext_vector_type(16))) __bf16 v16bf;
typedef __attribute__((ext_vector_type(8)))  float  v8f;

#define NJ 200000
#define NG 100000
#define NGP 100032          // grain rows padded to multiple of 64
#define EJJ 600000
#define EPG 600000
#define EGJ 600000

struct Ptr4 { const float* p[4]; };   // per gate block: 0=i 1=f 2=c 3=o

// ---------------- helpers ----------------

__device__ __forceinline__ unsigned short f2bf(float f) {
    union { float f; unsigned u; } v; v.f = f;
    unsigned u = v.u;
    unsigned r = (u + 0x7FFFu + ((u >> 16) & 1u)) >> 16;   // RNE
    return (unsigned short)r;
}

// inverse of the 16-bit WMMA operand k-mapping:
// lane<16 : e<8 -> k=e      ; e>=8 -> k=e+8   (k in 0..7, 16..23)
// lane>=16: e<8 -> k=e+8    ; e>=8 -> k=e+16  (k in 8..15, 24..31)
__device__ __forceinline__ void frag_pos(int kk, int lane_lo, int& lane, int& e) {
    if (kk < 8)       { lane = lane_lo;      e = kk;      }
    else if (kk < 16) { lane = lane_lo + 16; e = kk - 8;  }
    else if (kk < 24) { lane = lane_lo;      e = kk - 8;  }
    else              { lane = lane_lo + 16; e = kk - 16; }
}

__device__ __forceinline__ float sigmoidf_(float x) {
    return 1.0f / (1.0f + __expf(-x));
}

// ---------------- topology kernels (once per launch) ----------------

__global__ void count_edges_k(float* cnt, const int* __restrict__ dst, int E) {
    int e = blockIdx.x * blockDim.x + threadIdx.x;
    if (e < E) atomicAdd(&cnt[dst[e]], 1.0f);
}

__global__ void recip_k(float* r, int n) {
    int i = blockIdx.x * blockDim.x + threadIdx.x;
    if (i < n) r[i] = 1.0f / fmaxf(r[i], 1.0f);
}

// ---------------- per-cell: segment-sum scatter ----------------

__global__ void scatter_add_k(float* __restrict__ acc, const float* __restrict__ xsrc,
                              const int* __restrict__ src, const int* __restrict__ dst,
                              int E, int IN) {
    long idx = (long)blockIdx.x * blockDim.x + threadIdx.x;
    long total = (long)E * IN;
    if (idx >= total) return;
    int e = (int)(idx / IN);
    int f = (int)(idx % IN);
    atomicAdd(&acc[(long)dst[e] * IN + f], xsrc[(long)src[e] * IN + f]);
}

// ---------------- per-cell: A builders (write bf16 fragment layout) -----------

__global__ void build_a_joint_k(unsigned short* __restrict__ A,
                                const float* __restrict__ mjj, const float* __restrict__ rjj,
                                const float* __restrict__ mgj, const float* __restrict__ rgj,
                                const float* __restrict__ xin, const float* __restrict__ h,
                                int IN, int KP, int nrows, int nrowsPad) {
    long idx = (long)blockIdx.x * blockDim.x + threadIdx.x;
    int ppr = KP >> 1;                              // k-pairs per row
    long total = (long)nrowsPad * ppr;
    if (idx >= total) return;
    int row = (int)(idx / ppr);
    int k0  = (int)(idx % ppr) * 2;
    float v[2];
    #pragma unroll
    for (int q = 0; q < 2; ++q) {
        int k = k0 + q;
        float x = 0.0f;
        if (row < nrows) {
            if (k < IN)            x = mjj[(long)row * IN + k] * rjj[row];
            else if (k < 2 * IN)   x = mgj[(long)row * IN + (k - IN)] * rgj[row];
            else if (k < 3 * IN)   x = xin[(long)row * IN + (k - 2 * IN)];
            else if (k < 3 * IN + 64) x = h[(long)row * 64 + (k - 3 * IN)];
        }
        v[q] = x;
    }
    int rt = row >> 4, lr = row & 15, slab = k0 >> 5, kk = k0 & 31, lane, e;
    frag_pos(kk, lr, lane, e);
    unsigned out = (unsigned)f2bf(v[0]) | ((unsigned)f2bf(v[1]) << 16);
    *(unsigned*)(A + (((long)rt * (KP >> 5) + slab) * 32 + lane) * 16 + e) = out;
}

__global__ void build_a_grain_k(unsigned short* __restrict__ A,
                                const float* __restrict__ mpg, const float* __restrict__ rpg,
                                const float* __restrict__ xin, const float* __restrict__ h,
                                int IN, int KP, int nrows, int nrowsPad) {
    long idx = (long)blockIdx.x * blockDim.x + threadIdx.x;
    int ppr = KP >> 1;
    long total = (long)nrowsPad * ppr;
    if (idx >= total) return;
    int row = (int)(idx / ppr);
    int k0  = (int)(idx % ppr) * 2;
    float v[2];
    #pragma unroll
    for (int q = 0; q < 2; ++q) {
        int k = k0 + q;
        float x = 0.0f;
        if (row < nrows) {
            if (k < IN)              x = mpg[(long)row * IN + k] * rpg[row];
            else if (k < 2 * IN)     x = xin[(long)row * IN + (k - IN)];
            else if (k < 2 * IN + 64) x = h[(long)row * 64 + (k - 2 * IN)];
        }
        v[q] = x;
    }
    int rt = row >> 4, lr = row & 15, slab = k0 >> 5, kk = k0 & 31, lane, e;
    frag_pos(kk, lr, lane, e);
    unsigned out = (unsigned)f2bf(v[0]) | ((unsigned)f2bf(v[1]) << 16);
    *(unsigned*)(A + (((long)rt * (KP >> 5) + slab) * 32 + lane) * 16 + e) = out;
}

// ---------------- per-cell: weight / bias packing ----------------

__global__ void pack_w_joint_k(unsigned short* __restrict__ W,
                               Ptr4 wl_jj, Ptr4 wl_gj, Ptr4 wr_jj, Ptr4 wr_gj, Ptr4 wh,
                               int IN, int KP) {
    int idx = blockIdx.x * blockDim.x + threadIdx.x;
    if (idx >= KP * 256) return;
    int k = idx >> 8, n = idx & 255, b = n >> 6, j = n & 63;
    float v = 0.0f;
    if (k < IN)               v = wl_jj.p[b][k * 64 + j];
    else if (k < 2 * IN)      v = wl_gj.p[b][(k - IN) * 64 + j];
    else if (k < 3 * IN)      v = wr_jj.p[b][(k - 2 * IN) * 64 + j] + wr_gj.p[b][(k - 2 * IN) * 64 + j];
    else if (k < 3 * IN + 64) v = wh.p[b][(k - 3 * IN) * 64 + j];
    int slab = k >> 5, kk = k & 31, nt = n >> 4, nn = n & 15, lane, e;
    frag_pos(kk, nn, lane, e);
    W[((slab * 16 + nt) * 32 + lane) * 16 + e] = f2bf(v);
}

__global__ void pack_w_grain_k(unsigned short* __restrict__ W,
                               Ptr4 wl_pg, Ptr4 wr_pg, Ptr4 wh,
                               int IN, int KP) {
    int idx = blockIdx.x * blockDim.x + threadIdx.x;
    if (idx >= KP * 256) return;
    int k = idx >> 8, n = idx & 255, b = n >> 6, j = n & 63;
    float v = 0.0f;
    if (k < IN)               v = wl_pg.p[b][k * 64 + j];
    else if (k < 2 * IN)      v = wr_pg.p[b][(k - IN) * 64 + j];
    else if (k < 2 * IN + 64) v = wh.p[b][(k - 2 * IN) * 64 + j];
    int slab = k >> 5, kk = k & 31, nt = n >> 4, nn = n & 15, lane, e;
    frag_pos(kk, nn, lane, e);
    W[((slab * 16 + nt) * 32 + lane) * 16 + e] = f2bf(v);
}

__global__ void pack_bias_k(float* __restrict__ bias, Ptr4 b1, Ptr4 b2, Ptr4 b3, int n3) {
    int n = threadIdx.x;                  // 256 threads
    int b = n >> 6, j = n & 63;
    float v = b1.p[b][j] + b2.p[b][j];
    if (n3) v += b3.p[b][j];
    bias[n] = v;
}

// ---------------- fused WMMA GEMM + LSTM ----------------
// Block: 256 threads = 8 waves; computes 64 rows x 256 cols of pre-activation,
// then applies the LSTM gate math. Gate column blocks: 0-63=i 64-127=f
// 128-191=c(g~) 192-255=o.

template<int KP>
__global__ __launch_bounds__(256) void gemm_lstm_k(
        const unsigned short* __restrict__ A, const unsigned short* __restrict__ W,
        const float* __restrict__ bias, float* __restrict__ h_st,
        float* __restrict__ c_st, int nrows) {
    constexpr int KS = KP / 32;
    __shared__ float lds[64 * 256];

    const int tid  = threadIdx.x;
    const int lane = tid & 31;
    const int wave = tid >> 5;
    const int rg = wave >> 1;             // row group 0..3 (16 rows each)
    const int cg = wave & 1;              // col group 0..1 (128 cols each)
    const int rt = blockIdx.x * 4 + rg;   // 16-row tile index

    v8f acc[8];
    #pragma unroll
    for (int t = 0; t < 8; ++t)
        #pragma unroll
        for (int r = 0; r < 8; ++r) acc[t][r] = 0.0f;

    for (int s = 0; s < KS; ++s) {
        const v16bf a = *(const v16bf*)(A + (((long)rt * KS + s) * 32 + lane) * 16);
        const unsigned short* wp = W + (((s * 16) + cg * 8) * 32 + lane) * 16;
        #pragma unroll
        for (int t = 0; t < 8; ++t) {
            const v16bf b = *(const v16bf*)(wp + (long)t * 512);
            acc[t] = __builtin_amdgcn_wmma_f32_16x16x32_bf16(
                false, a, false, b, (short)0, acc[t], false, false);
        }
    }

    // scatter accumulators to LDS (C layout: lane n = lane&15, row = r + 8*(lane>=16))
    const int mhalf = (lane >> 4) << 3;
    const int nlo   = lane & 15;
    #pragma unroll
    for (int t = 0; t < 8; ++t)
        #pragma unroll
        for (int r = 0; r < 8; ++r)
            lds[(rg * 16 + r + mhalf) * 256 + (cg * 8 + t) * 16 + nlo] = acc[t][r];
    __syncthreads();

    // LSTM: 64 rows x 64 units
    #pragma unroll
    for (int it = 0; it < 16; ++it) {
        int idx = tid + it * 256;
        int row = idx >> 6, u = idx & 63;
        int grow = blockIdx.x * 64 + row;
        if (grow < nrows) {
            float pi = lds[row * 256 +       u] + bias[u];
            float pf = lds[row * 256 +  64 + u] + bias[64 + u];
            float pc = lds[row * 256 + 128 + u] + bias[128 + u];
            float po = lds[row * 256 + 192 + u] + bias[192 + u];
            float ig = sigmoidf_(pi);
            float fg = sigmoidf_(pf);
            float gg = tanhf(pc);
            float og = sigmoidf_(po);
            long  o  = (long)grow * 64 + u;
            float cn = fg * c_st[o] + ig * gg;
            c_st[o] = cn;
            h_st[o] = og * tanhf(cn);
        }
    }
}

// ---------------- edge head ----------------

__global__ void edge_head_k(const float* __restrict__ h,
                            const int* __restrict__ src, const int* __restrict__ dst,
                            const float* __restrict__ ea,
                            const float* __restrict__ w1, const float* __restrict__ b1,
                            const float* __restrict__ w2, const float* __restrict__ b2,
                            float* __restrict__ out_event, float* __restrict__ out_edge, int E) {
    int e = blockIdx.x * blockDim.x + threadIdx.x;
    if (e >= E) return;
    const float* hs = h + (long)src[e] * 64;
    const float* hd = h + (long)dst[e] * 64;
    float d0 = 0.f, d1 = 0.f, d2 = 0.f;
    #pragma unroll 4
    for (int k = 0; k < 64; ++k) {
        float a = hs[k];
        d0 += a * w1[k * 2 + 0]; d1 += a * w1[k * 2 + 1]; d2 += a * w2[k];
    }
    #pragma unroll 4
    for (int k = 0; k < 64; ++k) {
        float a = hd[k];
        d0 += a * w1[(64 + k) * 2 + 0]; d1 += a * w1[(64 + k) * 2 + 1]; d2 += a * w2[64 + k];
    }
    float a = ea[e];
    d0 += a * w1[128 * 2 + 0]; d1 += a * w1[128 * 2 + 1]; d2 += a * w2[128];
    out_event[e]        = d2 + b2[0];
    out_edge[2 * e + 0] = tanhf(d0 + b1[0]);
    out_edge[2 * e + 1] = tanhf(d1 + b1[1]);
}

// ---------------- host orchestration ----------------

static inline int gridFor(long total, int bs) { return (int)((total + bs - 1) / bs); }

extern "C" void kernel_launch(void* const* d_in, const int* in_sizes, int n_in,
                              void* d_out, int out_size, void* d_ws, size_t ws_size,
                              hipStream_t stream) {
    (void)in_sizes; (void)n_in; (void)out_size; (void)ws_size;

    auto P = [&](int i) -> const float* { return (const float*)d_in[i]; };

    const float* x_joint = P(0);
    const float* x_grain = P(1);
    const int* ei_jj = (const int*)d_in[2];
    const int* ei_pg = (const int*)d_in[3];
    const int* ei_gj = (const int*)d_in[4];
    const float* ea_jj = P(5);

    // params: jax sorted-key flatten -> dec[0], dec[1], enc[0], enc[1], lin1{b,w}, lin2{b,w}
    // cell leaves: bh(8: g{c,f,i,o} x nt{grain,joint}), conv(36: g x et{gj,jj,pg} x {b,wl,wr}), wh(8)
    const int BASE = 6;
    const int cellBaseExec[4] = { BASE + 104, BASE + 156, BASE + 0, BASE + 52 }; // enc0,enc1,dec0,dec1
    const int GS[4] = { 2, 1, 0, 3 };  // gate block (i,f,c,o) -> sorted-key index (c,f,i,o)
    const float* lin1_b = P(BASE + 208);
    const float* lin1_w = P(BASE + 209);
    const float* lin2_b = P(BASE + 210);
    const float* lin2_w = P(BASE + 211);

    // ---- workspace carve-up ----
    size_t off = 0;
    auto alloc = [&](size_t bytes) -> void* {
        off = (off + 255) & ~(size_t)255;
        void* p = (char*)d_ws + off;
        off += bytes;
        return p;
    };
    float* recip_jj = (float*)alloc((size_t)NJ * 4);
    float* recip_gj = (float*)alloc((size_t)NJ * 4);
    float* recip_pg = (float*)alloc((size_t)NG * 4);
    float* m_jj = (float*)alloc((size_t)NJ * 64 * 4);
    float* m_gj = (float*)alloc((size_t)NJ * 64 * 4);
    float* m_pg = (float*)alloc((size_t)NG * 64 * 4);
    unsigned short* A_j = (unsigned short*)alloc((size_t)NJ  * 256 * 2);
    unsigned short* A_g = (unsigned short*)alloc((size_t)NGP * 192 * 2);
    unsigned short* W_j = (unsigned short*)alloc((size_t)256 * 256 * 2);
    unsigned short* W_g = (unsigned short*)alloc((size_t)192 * 256 * 2);
    float* bias_j = (float*)alloc(256 * 4);
    float* bias_g = (float*)alloc(256 * 4);
    float* h0j = (float*)alloc((size_t)NJ  * 64 * 4);
    float* h0g = (float*)alloc((size_t)NGP * 64 * 4);
    float* c0j = (float*)alloc((size_t)NJ  * 64 * 4);
    float* c0g = (float*)alloc((size_t)NGP * 64 * 4);
    float* h1j = (float*)alloc((size_t)NJ  * 64 * 4);
    float* h1g = (float*)alloc((size_t)NGP * 64 * 4);
    float* c1j = (float*)alloc((size_t)NJ  * 64 * 4);
    float* c1g = (float*)alloc((size_t)NGP * 64 * 4);

    // ---- init state + counts (deterministic each call) ----
    hipMemsetAsync(h0j, 0, (size_t)NJ  * 64 * 4, stream);
    hipMemsetAsync(h0g, 0, (size_t)NGP * 64 * 4, stream);
    hipMemsetAsync(c0j, 0, (size_t)NJ  * 64 * 4, stream);
    hipMemsetAsync(c0g, 0, (size_t)NGP * 64 * 4, stream);
    hipMemsetAsync(h1j, 0, (size_t)NJ  * 64 * 4, stream);
    hipMemsetAsync(h1g, 0, (size_t)NGP * 64 * 4, stream);
    hipMemsetAsync(c1j, 0, (size_t)NJ  * 64 * 4, stream);
    hipMemsetAsync(c1g, 0, (size_t)NGP * 64 * 4, stream);
    hipMemsetAsync(recip_jj, 0, (size_t)NJ * 4, stream);
    hipMemsetAsync(recip_gj, 0, (size_t)NJ * 4, stream);
    hipMemsetAsync(recip_pg, 0, (size_t)NG * 4, stream);

    count_edges_k<<<gridFor(EJJ, 256), 256, 0, stream>>>(recip_jj, ei_jj + EJJ, EJJ);
    count_edges_k<<<gridFor(EGJ, 256), 256, 0, stream>>>(recip_gj, ei_gj + EGJ, EGJ);
    count_edges_k<<<gridFor(EPG, 256), 256, 0, stream>>>(recip_pg, ei_pg + EPG, EPG);
    recip_k<<<gridFor(NJ, 256), 256, 0, stream>>>(recip_jj, NJ);
    recip_k<<<gridFor(NJ, 256), 256, 0, stream>>>(recip_gj, NJ);
    recip_k<<<gridFor(NG, 256), 256, 0, stream>>>(recip_pg, NG);

    // ---- 4 cells: enc0, enc1, dec0, dec1 ----
    for (int ci = 0; ci < 4; ++ci) {
        const int B  = cellBaseExec[ci];
        const int IN = (ci == 0 || ci == 2) ? 5 : 64;
        const int KPj = (IN == 5) ? 96 : 256;
        const int KPg = (IN == 5) ? 96 : 192;
        const float* xin_j = (ci == 0 || ci == 2) ? x_joint : h0j;
        const float* xin_g = (ci == 0 || ci == 2) ? x_grain : h0g;
        float* hj = (ci == 0 || ci == 2) ? h0j : h1j;
        float* hg = (ci == 0 || ci == 2) ? h0g : h1g;
        float* cj = (ci == 0 || ci == 2) ? c0j : c1j;
        float* cgp = (ci == 0 || ci == 2) ? c0g : c1g;

        // gather per-gate param pointers (block order i,f,c,o)
        Ptr4 wl_jj, wr_jj, b_jj, wl_gj, wr_gj, b_gj, wl_pg, wr_pg, b_pg;
        Ptr4 wh_j, wh_g, bh_j, bh_g;
        for (int b = 0; b < 4; ++b) {
            int g = GS[b];
            // conv: base + 8 + g*9 + et*3 + leaf  (et: gj=0 jj=1 pg=2; leaf: b=0 wl=1 wr=2)
            b_gj.p[b]  = P(B + 8 + g * 9 + 0 * 3 + 0);
            wl_gj.p[b] = P(B + 8 + g * 9 + 0 * 3 + 1);
            wr_gj.p[b] = P(B + 8 + g * 9 + 0 * 3 + 2);
            b_jj.p[b]  = P(B + 8 + g * 9 + 1 * 3 + 0);
            wl_jj.p[b] = P(B + 8 + g * 9 + 1 * 3 + 1);
            wr_jj.p[b] = P(B + 8 + g * 9 + 1 * 3 + 2);
            b_pg.p[b]  = P(B + 8 + g * 9 + 2 * 3 + 0);
            wl_pg.p[b] = P(B + 8 + g * 9 + 2 * 3 + 1);
            wr_pg.p[b] = P(B + 8 + g * 9 + 2 * 3 + 2);
            bh_g.p[b] = P(B + g * 2 + 0);      // nt sorted: grain, joint
            bh_j.p[b] = P(B + g * 2 + 1);
            wh_g.p[b] = P(B + 44 + g * 2 + 0);
            wh_j.p[b] = P(B + 44 + g * 2 + 1);
        }

        // 1. segment sums
        hipMemsetAsync(m_jj, 0, (size_t)NJ * IN * 4, stream);
        hipMemsetAsync(m_gj, 0, (size_t)NJ * IN * 4, stream);
        hipMemsetAsync(m_pg, 0, (size_t)NG * IN * 4, stream);
        scatter_add_k<<<gridFor((long)EJJ * IN, 256), 256, 0, stream>>>(
            m_jj, xin_j, ei_jj, ei_jj + EJJ, EJJ, IN);
        scatter_add_k<<<gridFor((long)EGJ * IN, 256), 256, 0, stream>>>(
            m_gj, xin_g, ei_gj, ei_gj + EGJ, EGJ, IN);
        scatter_add_k<<<gridFor((long)EPG * IN, 256), 256, 0, stream>>>(
            m_pg, xin_j, ei_pg, ei_pg + EPG, EPG, IN);

        // 2. A builders (fragment-layout bf16)
        build_a_joint_k<<<gridFor((long)NJ * (KPj >> 1), 256), 256, 0, stream>>>(
            A_j, m_jj, recip_jj, m_gj, recip_gj, xin_j, hj, IN, KPj, NJ, NJ);
        build_a_grain_k<<<gridFor((long)NGP * (KPg >> 1), 256), 256, 0, stream>>>(
            A_g, m_pg, recip_pg, xin_g, hg, IN, KPg, NG, NGP);

        // 3. weight / bias packing
        pack_w_joint_k<<<gridFor((long)KPj * 256, 256), 256, 0, stream>>>(
            W_j, wl_jj, wl_gj, wr_jj, wr_gj, wh_j, IN, KPj);
        pack_w_grain_k<<<gridFor((long)KPg * 256, 256), 256, 0, stream>>>(
            W_g, wl_pg, wr_pg, wh_g, IN, KPg);
        pack_bias_k<<<1, 256, 0, stream>>>(bias_j, b_jj, b_gj, bh_j, 1);
        pack_bias_k<<<1, 256, 0, stream>>>(bias_g, b_pg, bh_g, bh_g, 0);

        // 4. fused WMMA GEMM + LSTM (in-place state update)
        if (KPj == 96)
            gemm_lstm_k<96><<<NJ / 64, 256, 0, stream>>>(A_j, W_j, bias_j, hj, cj, NJ);
        else
            gemm_lstm_k<256><<<NJ / 64, 256, 0, stream>>>(A_j, W_j, bias_j, hj, cj, NJ);
        if (KPg == 96)
            gemm_lstm_k<96><<<NGP / 64, 256, 0, stream>>>(A_g, W_g, bias_g, hg, cgp, NG);
        else
            gemm_lstm_k<192><<<NGP / 64, 256, 0, stream>>>(A_g, W_g, bias_g, hg, cgp, NG);
    }

    // ---- edge head on final joint hidden state (h1j) ----
    float* out_event = (float*)d_out;
    float* out_edge  = out_event + EJJ;
    edge_head_k<<<gridFor(EJJ, 256), 256, 0, stream>>>(
        h1j, ei_jj, ei_jj + EJJ, ea_jj, lin1_w, lin1_b, lin2_w, lin2_b,
        out_event, out_edge, EJJ);
}
